// EdgeAtt_58042188038337
// MI455X (gfx1250) — matrix operand order repaired and checked by reading, not compiled, not run on hardware
//
#include <hip/hip_runtime.h>

typedef __attribute__((ext_vector_type(16))) __bf16 v16bf;
typedef __attribute__((ext_vector_type(8)))  __bf16 v8bf;
typedef __attribute__((ext_vector_type(4)))  __bf16 v4bf;
typedef __attribute__((ext_vector_type(8)))  float  v8f;

#define BATCH 1024
#define LSEQ  110
#define LPAD  112
#define GD    512
#define NF_STRIDE 520                                  // bf16 elements, stride%64banks==4 -> conflict-free
#define NF_OFF 0
#define ATT_OFF (LPAD * NF_STRIDE * 2)                 // 116480 B
#define SC_OFF  (ATT_OFF + LPAD * NF_STRIDE * 2)       // 232960 B
#define SC_STRIDE 116                                  // f32 elements
#define SMEM_BYTES (SC_OFF + LPAD * SC_STRIDE * 4)     // 284928 B  (<= 320KB/WG on CDNA5)

union Frag { v16bf v; v8bf h[2]; };

// ---- Kernel 1: convert weight f32 -> bf16 once (shared by all batches, L2-resident) ----
__global__ void w_to_bf16_kernel(const float* __restrict__ w, __bf16* __restrict__ wb) {
    int i = blockIdx.x * blockDim.x + threadIdx.x;
    if (i < GD * GD) wb[i] = (__bf16)w[i];
}

// ---- Kernel 2: fused att GEMM + banded scores GEMM + masked softmax, one block per batch ----
__global__ __launch_bounds__(256)
void edge_att_kernel(const float* __restrict__ nf,
                     const __bf16* __restrict__ wb,
                     const int* __restrict__ tlen,
                     float* __restrict__ out) {
    extern __shared__ __align__(16) char smem[];
    const int b    = blockIdx.x;
    const int tid  = threadIdx.x;
    const int wave = tid >> 5;
    const int lane = tid & 31;
    const int r    = lane & 15;       // row-in-tile / column lane
    const int hf   = lane >> 4;       // half-wave select

    __bf16* nfs = (__bf16*)(smem + NF_OFF);
    __bf16* ats = (__bf16*)(smem + ATT_OFF);
    float*  scs = (float*) (smem + SC_OFF);

    // ---------- Phase 0: stage nf[b] (f32) into LDS as bf16; zero pad rows 110..111 ----------
    {
        const float4* src = (const float4*)(nf + (size_t)b * LSEQ * GD);
        for (int idx = tid; idx < LSEQ * (GD / 4); idx += 256) {
            int row = idx >> 7;            // 128 float4 per row
            int c4  = idx & 127;
            float4 x = src[idx];
            v4bf pv = { (__bf16)x.x, (__bf16)x.y, (__bf16)x.z, (__bf16)x.w };
            *(v4bf*)(nfs + row * NF_STRIDE + c4 * 4) = pv;
        }
        // pad rows: 2 rows * 128 float4-slots = 256 slots, exactly one per thread
        {
            v4bf z = {};
            int row = 110 + (tid >> 7);
            int c4  = tid & 127;
            *(v4bf*)(nfs + row * NF_STRIDE + c4 * 4) = z;
        }
    }
    __syncthreads();

    // ---------- Phase 1: att = nf @ W^T  (bf16 WMMA, f32 accum) ----------
    // Each wave owns n-tiles {wave, wave+8, wave+16, wave+24}. For each, cache the full
    // K=512 of B-fragments (W^T) in registers (16 frags x 8 VGPRs), then sweep 7 M-tiles.
    for (int nn = wave; nn < 32; nn += 8) {
        Frag bw[16];
        const __bf16* wrow = wb + (size_t)(nn * 16 + r) * GD + hf * 8;
#pragma unroll
        for (int k = 0; k < 16; ++k) {
            bw[k].h[0] = *(const v8bf*)(wrow + k * 32);
            bw[k].h[1] = *(const v8bf*)(wrow + k * 32 + 16);
        }
        for (int m = 0; m < 7; ++m) {
            v8f acc = {};
            const __bf16* arow = nfs + (m * 16 + r) * NF_STRIDE + hf * 8;
#pragma unroll
            for (int k = 0; k < 16; ++k) {
                Frag a;
                a.h[0] = *(const v8bf*)(arow + k * 32);        // K = 32k + 8*hf   .. +7
                a.h[1] = *(const v8bf*)(arow + k * 32 + 16);   // K = 32k+16+8*hf .. +7
                acc = __builtin_amdgcn_wmma_f32_16x16x32_bf16(
                        false, a.v, false, bw[k].v, (short)0, acc, false, false);
            }
            // D layout: VGPR i, lane -> row = m*16 + i + 8*hf, col = nn*16 + r
            __bf16* drow = ats + nn * 16 + r;
#pragma unroll
            for (int i = 0; i < 8; ++i)
                drow[(m * 16 + i + hf * 8) * NF_STRIDE] = (__bf16)acc[i];
        }
    }
    __syncthreads();

    // ---------- Phase 2: banded scores[j,l] = nf[j,:] . att[l,:]  (19 of 49 tiles) ----------
    for (int t = wave; t < 19; t += 8) {
        const int J = (t + 1) / 3;            // j-tile
        const int L = J - 1 + (t + 1) % 3;    // l-tile in {J-1, J, J+1}
        v8f acc = {};
        const __bf16* arow = nfs + (J * 16 + r) * NF_STRIDE + hf * 8;
        const __bf16* brow = ats + (L * 16 + r) * NF_STRIDE + hf * 8;
#pragma unroll
        for (int k = 0; k < 16; ++k) {
            Frag a, bb;
            a.h[0]  = *(const v8bf*)(arow + k * 32);
            a.h[1]  = *(const v8bf*)(arow + k * 32 + 16);
            bb.h[0] = *(const v8bf*)(brow + k * 32);
            bb.h[1] = *(const v8bf*)(brow + k * 32 + 16);
            acc = __builtin_amdgcn_wmma_f32_16x16x32_bf16(
                    false, a.v, false, bb.v, (short)0, acc, false, false);
        }
        float* drow = scs + L * 16 + r;
#pragma unroll
        for (int i = 0; i < 8; ++i)
            drow[(J * 16 + i + hf * 8) * SC_STRIDE] = acc[i];
    }
    __syncthreads();

    // ---------- Phase 3: banded masked softmax (one row per thread) ----------
    const int len = tlen[b];
    if (tid < LSEQ) {
        const int j = tid;
        if (j < len) {
            int lo = j - 10; if (lo < 0) lo = 0;
            int hi = j + 10; if (hi > len - 1) hi = len - 1;
            float* row = scs + j * SC_STRIDE;
            float mx = -3.4e38f;
            for (int l = lo; l <= hi; ++l) mx = fmaxf(mx, row[l]);
            float s = 0.f;
            for (int l = lo; l <= hi; ++l) { float e = __expf(row[l] - mx); row[l] = e; s += e; }
            float inv = 1.0f / s;
            for (int l = lo; l <= hi; ++l) row[l] *= inv;
        }
    }
    __syncthreads();

    // ---------- Phase 4: coalesced masked write of the full 110x110 tile ----------
    float* ob = out + (size_t)b * (LSEQ * LSEQ);
    for (int idx = tid; idx < LSEQ * LSEQ; idx += 256) {
        int j = idx / LSEQ;
        int l = idx - j * LSEQ;
        bool ok = (j < len) && (l < len) && (l >= j - 10) && (l <= j + 10);
        ob[idx] = ok ? scs[j * SC_STRIDE + l] : 0.0f;
    }
}

extern "C" void kernel_launch(void* const* d_in, const int* in_sizes, int n_in,
                              void* d_out, int out_size, void* d_ws, size_t ws_size,
                              hipStream_t stream) {
    (void)in_sizes; (void)n_in; (void)out_size; (void)ws_size;
    const float* nf = (const float*)d_in[0];   // (1024, 110, 512) f32
    const float* w  = (const float*)d_in[1];   // (512, 512) f32
    const int*   tl = (const int*)d_in[2];     // (1024,) int32
    float* out = (float*)d_out;                // (1024, 110, 110) f32
    __bf16* wb = (__bf16*)d_ws;                // 512 KB bf16 weight

    (void)hipFuncSetAttribute((const void*)edge_att_kernel,
                              hipFuncAttributeMaxDynamicSharedMemorySize, SMEM_BYTES);

    w_to_bf16_kernel<<<(GD * GD) / 256, 256, 0, stream>>>(w, wb);
    edge_att_kernel<<<BATCH, 256, SMEM_BYTES, stream>>>(nf, wb, tl, out);
}
